// DFTModel_41532333753051
// MI455X (gfx1250) — compile-verified, matrix operand order checked by reading, not run on hardware
//
#include <hip/hip_runtime.h>
#include <hip/hip_bf16.h>

typedef __attribute__((ext_vector_type(16))) _Float16 v16h;
typedef __attribute__((ext_vector_type(8)))  float    v8f;
typedef __attribute__((ext_vector_type(4)))  unsigned int v4u;

#define IMG 222
#define BATCH 256
#define NCOEFF 668   // 111 + 4*56 + 9*37
#define KPAD0 672    // 668 padded to multiple of 32
#define N5PAD 1024   // 1000 padded to multiple of 64

__device__ __forceinline__ int imin(int a, int b) { return a < b ? a : b; }

// ---------------------------------------------------------------------------
// Kernel 1: one workgroup per batch image. Lane = image ROW, loop over
// columns: the ring index d = min(d_row, d_col) is then distinct across
// lanes almost everywhere (d_row differs per lane), so the ds_add_f32 bin
// updates are nearly conflict-free instead of 32-way same-address.
// Strided loads are L0-resident (≈16 KB live lines per CU, reused for ~10
// consecutive columns), so HBM traffic remains one pass over x.
// ---------------------------------------------------------------------------
__global__ __launch_bounds__(256) void ring_sums_kernel(
    const float* __restrict__ x, float* __restrict__ rs)
{
    __shared__ float sm[NCOEFF];
    const int b = blockIdx.x;
    const int t = threadIdx.x;
    for (int k = t; k < NCOEFF; k += 256) sm[k] = 0.f;
    __syncthreads();

    if (t < IMG) {
        const int i = t;
        // per-row (per-lane) components, hoisted out of the column loop
        const int d1i = imin(i, 221 - i);
        const int bi2 = (i >= 111) ? 1 : 0;
        const int ri2 = i - bi2 * 111;
        const int d2i = imin(ri2, 110 - ri2);
        const int bi3 = i / 74;
        const int ri3 = i - bi3 * 74;
        const int d3i = imin(ri3, 73 - ri3);
        const int base2 = 111 + bi2 * 2 * 56;
        const int base3 = 335 + bi3 * 3 * 37;

        const float* xr = x + ((size_t)b * IMG + i) * IMG * 3;
        for (int j = 0; j < IMG; ++j) {
            __builtin_prefetch(xr + (j + 16) * 3, 0, 0);
            const float v0 = xr[j * 3 + 0];
            const float v1 = xr[j * 3 + 1];
            const float v2 = xr[j * 3 + 2];
            // level 1: bs = 222, rings 0..110 at offset 0
            {
                const int d1j = imin(j, 221 - j);
                atomicAdd(&sm[imin(d1i, d1j)], v0);
            }
            // level 2: bs = 111, rings 0..55, blocks row-major, offset 111
            {
                const int bj2 = (j >= 111) ? 1 : 0;
                const int rj2 = j - bj2 * 111;
                const int d2j = imin(rj2, 110 - rj2);
                atomicAdd(&sm[base2 + bj2 * 56 + imin(d2i, d2j)], v1);
            }
            // level 3: bs = 74, rings 0..36, blocks row-major, offset 335
            {
                const int bj3 = j / 74;
                const int rj3 = j - bj3 * 74;
                const int d3j = imin(rj3, 73 - rj3);
                atomicAdd(&sm[base3 + bj3 * 37 + imin(d3i, d3j)], v2);
            }
        }
    }
    __syncthreads();
    for (int k = t; k < NCOEFF; k += 256) rs[(size_t)b * NCOEFF + k] = sm[k];
}

// ---------------------------------------------------------------------------
// Kernel 2: coeff(r) = sum_{d>=r} w[n-1-d] * ringsum(d)  (suffix scan).
// 14 blocks per batch: 1 (lvl1) + 4 (lvl2) + 9 (lvl3).
// ---------------------------------------------------------------------------
__global__ void coeff_kernel(const float* __restrict__ rs,
                             const float* __restrict__ w1,
                             const float* __restrict__ w2,
                             const float* __restrict__ w3,
                             float* __restrict__ c)
{
    const int b = blockIdx.x;
    const int t = threadIdx.x;
    if (t >= 14) return;
    int base, n;
    const float* w;
    if (t == 0)      { base = 0;                  n = 111; w = w1; }
    else if (t < 5)  { int blk = t - 1; base = 111 + blk * 56; n = 56; w = w2 + blk * 56; }
    else             { int blk = t - 5; base = 335 + blk * 37; n = 37; w = w3 + blk * 37; }
    const float* r  = rs + (size_t)b * NCOEFF + base;
    float*       co = c  + (size_t)b * NCOEFF + base;
    float acc = 0.f;
    for (int d = n - 1; d >= 0; --d) {
        acc += w[n - 1 - d] * r[d];
        co[d] = acc;
    }
}

// ---------------------------------------------------------------------------
// Kernel 3: per-feature batch stats (biased var) -> fused scale/shift.
// ---------------------------------------------------------------------------
__global__ __launch_bounds__(256) void bn_stats_kernel(
    const float* __restrict__ c, const float* __restrict__ gamma,
    const float* __restrict__ beta, float* __restrict__ scale,
    float* __restrict__ shift)
{
    const int k = blockIdx.x;
    const int t = threadIdx.x;
    __shared__ float s1[256];
    __shared__ float s2[256];
    float v = c[(size_t)t * NCOEFF + k];
    s1[t] = v;
    s2[t] = v * v;
    __syncthreads();
    for (int s = 128; s > 0; s >>= 1) {
        if (t < s) { s1[t] += s1[t + s]; s2[t] += s2[t + s]; }
        __syncthreads();
    }
    if (t == 0) {
        float mu   = s1[0] * (1.f / 256.f);
        float var  = s2[0] * (1.f / 256.f) - mu * mu;
        float rstd = rsqrtf(var + 1e-5f);
        float sc   = gamma[k] * rstd;
        scale[k] = sc;
        shift[k] = beta[k] - mu * sc;
    }
}

// ---------------------------------------------------------------------------
// Kernel 4: normalized coeffs -> f16 activation matrix [256][672] (zero pad).
// ---------------------------------------------------------------------------
__global__ void pack_h0_kernel(const float* __restrict__ c,
                               const float* __restrict__ scale,
                               const float* __restrict__ shift,
                               _Float16* __restrict__ h0)
{
    const int idx = blockIdx.x * 256 + threadIdx.x;
    if (idx >= BATCH * KPAD0) return;
    const int b = idx / KPAD0;
    const int k = idx - b * KPAD0;
    float v = 0.f;
    if (k < NCOEFF) v = c[(size_t)b * NCOEFF + k] * scale[k] + shift[k];
    h0[idx] = (_Float16)v;
}

// ---------------------------------------------------------------------------
// Kernel 5: f32 weights -> f16 with zero padding to (Npad, Kpad).
// ---------------------------------------------------------------------------
__global__ void convert_w_kernel(const float* __restrict__ W,
                                 _Float16* __restrict__ Wh,
                                 int N, int K, int Npad, int Kpad)
{
    const size_t idx = (size_t)blockIdx.x * 256 + threadIdx.x;
    const size_t tot = (size_t)Npad * Kpad;
    if (idx >= tot) return;
    const int n = (int)(idx / Kpad);
    const int k = (int)(idx - (size_t)n * Kpad);
    Wh[idx] = (n < N && k < K) ? (_Float16)W[(size_t)n * K + k] : (_Float16)0.f;
}

// ---------------------------------------------------------------------------
// Kernel 6: wave32 WMMA GEMM, 16x64 output strip per wave (4 accumulators).
//   A: [M][K] f16 row-major (activations), B: [Npad][K] f16 row-major
//   (weights, so B row n is the K-vector for output feature n).
// One A fragment is reused across 4 back-to-back v_wmma issues per K-step:
// 10 b128 loads / 4 WMMAs instead of 4 / 1.
// Per-lane fragment loads follow the ISA 16-bit A layout:
//   lane L -> row L%16, half = L/16; set s in {0,1}:
//   halves k = k0 + s*16 + half*8 .. +7  => two contiguous 16B loads.
// C/D layout: VGPR i -> M = i + 8*(L/16), N = L%16.
// ---------------------------------------------------------------------------
__global__ __launch_bounds__(32) void wmma_gemm_kernel(
    const _Float16* __restrict__ A, const _Float16* __restrict__ B,
    const float* __restrict__ bias, _Float16* __restrict__ outH,
    float* __restrict__ outF, int K, int ldo, int Nvalid, int relu)
{
    const int tileN4 = blockIdx.x;   // 64-wide N strip
    const int tileM  = blockIdx.y;
    const int lane   = threadIdx.x;
    const int half   = lane >> 4;
    const int l16    = lane & 15;

    const _Float16* arow = A + (size_t)(tileM * 16 + l16) * K;
    const _Float16* brow[4];
#pragma unroll
    for (int s = 0; s < 4; ++s)
        brow[s] = B + (size_t)(tileN4 * 64 + s * 16 + l16) * K;

    v8f acc[4] = {{0.f, 0.f, 0.f, 0.f, 0.f, 0.f, 0.f, 0.f},
                  {0.f, 0.f, 0.f, 0.f, 0.f, 0.f, 0.f, 0.f},
                  {0.f, 0.f, 0.f, 0.f, 0.f, 0.f, 0.f, 0.f},
                  {0.f, 0.f, 0.f, 0.f, 0.f, 0.f, 0.f, 0.f}};

#pragma unroll 2
    for (int k0 = 0; k0 < K; k0 += 32) {
        union { v16h h; v4u u[2]; } af;
        af.u[0] = *(const v4u*)(arow + k0 + half * 8);
        af.u[1] = *(const v4u*)(arow + k0 + 16 + half * 8);
        __builtin_prefetch(arow + k0 + 64 + half * 8, 0, 0);
#pragma unroll
        for (int s = 0; s < 4; ++s) {
            union { v16h h; v4u u[2]; } bf;
            bf.u[0] = *(const v4u*)(brow[s] + k0 + half * 8);
            bf.u[1] = *(const v4u*)(brow[s] + k0 + 16 + half * 8);
            acc[s] = __builtin_amdgcn_wmma_f32_16x16x32_f16(
                /*neg_a=*/false, af.h, /*neg_b=*/false, bf.h,
                /*c_mod=*/(short)0, acc[s], /*reuse_a=*/false, /*reuse_b=*/false);
        }
    }

#pragma unroll
    for (int s = 0; s < 4; ++s) {
        const int n = tileN4 * 64 + s * 16 + l16;
        const float bval = (n < Nvalid) ? bias[n] : 0.f;
#pragma unroll
        for (int i = 0; i < 8; ++i) {
            const int m = tileM * 16 + i + half * 8;
            float v = acc[s][i] + bval;
            if (relu) v = fmaxf(v, 0.f);
            if (outH) outH[(size_t)m * ldo + n] = (_Float16)v;
            if (outF && n < Nvalid) outF[(size_t)m * ldo + n] = v;
        }
    }
}

// ---------------------------------------------------------------------------
extern "C" void kernel_launch(void* const* d_in, const int* in_sizes, int n_in,
                              void* d_out, int out_size, void* d_ws, size_t ws_size,
                              hipStream_t stream)
{
    const float* x     = (const float*)d_in[0];
    const float* w1    = (const float*)d_in[1];
    const float* w2    = (const float*)d_in[2];
    const float* w3    = (const float*)d_in[3];
    const float* gamma = (const float*)d_in[4];
    const float* beta  = (const float*)d_in[5];
    const float* fc1w  = (const float*)d_in[6];
    const float* fc1b  = (const float*)d_in[7];
    const float* fc2w  = (const float*)d_in[8];
    const float* fc2b  = (const float*)d_in[9];
    const float* fc3w  = (const float*)d_in[10];
    const float* fc3b  = (const float*)d_in[11];
    const float* fc4w  = (const float*)d_in[12];
    const float* fc4b  = (const float*)d_in[13];
    const float* fc5w  = (const float*)d_in[14];
    const float* fc5b  = (const float*)d_in[15];
    float* out = (float*)d_out;

    // ---- workspace partition (all chunks multiples of 128 bytes) ----
    char* p = (char*)d_ws;
    float* rs    = (float*)p;           p += (size_t)BATCH * NCOEFF * 4;   // 684032
    float* c     = (float*)p;           p += (size_t)BATCH * NCOEFF * 4;   // 684032
    float* scale = (float*)p;           p += 2688;
    float* shift = (float*)p;           p += 2688;
    _Float16* h0 = (_Float16*)p;        p += (size_t)BATCH * KPAD0 * 2;    // 344064
    _Float16* h1 = (_Float16*)p;        p += (size_t)BATCH * 2048 * 2;
    _Float16* h2 = (_Float16*)p;        p += (size_t)BATCH * 1024 * 2;
    _Float16* h3 = (_Float16*)p;        p += (size_t)BATCH * 512 * 2;
    _Float16* h4 = (_Float16*)p;        p += (size_t)BATCH * 128 * 2;
    _Float16* W1h = (_Float16*)p;       p += (size_t)2048 * KPAD0 * 2;
    _Float16* W2h = (_Float16*)p;       p += (size_t)1024 * 2048 * 2;
    _Float16* W3h = (_Float16*)p;       p += (size_t)512 * 1024 * 2;
    _Float16* W4h = (_Float16*)p;       p += (size_t)128 * 512 * 2;
    _Float16* W5h = (_Float16*)p;       p += (size_t)N5PAD * 128 * 2;

    // ---- feature extraction ----
    ring_sums_kernel<<<BATCH, 256, 0, stream>>>(x, rs);
    coeff_kernel<<<BATCH, 32, 0, stream>>>(rs, w1, w2, w3, c);
    bn_stats_kernel<<<NCOEFF, 256, 0, stream>>>(c, gamma, beta, scale, shift);
    pack_h0_kernel<<<(BATCH * KPAD0 + 255) / 256, 256, 0, stream>>>(c, scale, shift, h0);

    // ---- weight conversion (f32 -> padded f16) ----
    {
        size_t t;
        t = (size_t)2048 * KPAD0;
        convert_w_kernel<<<(unsigned)((t + 255) / 256), 256, 0, stream>>>(fc1w, W1h, 2048, NCOEFF, 2048, KPAD0);
        t = (size_t)1024 * 2048;
        convert_w_kernel<<<(unsigned)((t + 255) / 256), 256, 0, stream>>>(fc2w, W2h, 1024, 2048, 1024, 2048);
        t = (size_t)512 * 1024;
        convert_w_kernel<<<(unsigned)((t + 255) / 256), 256, 0, stream>>>(fc3w, W3h, 512, 1024, 512, 1024);
        t = (size_t)128 * 512;
        convert_w_kernel<<<(unsigned)((t + 255) / 256), 256, 0, stream>>>(fc4w, W4h, 128, 512, 128, 512);
        t = (size_t)N5PAD * 128;
        convert_w_kernel<<<(unsigned)((t + 255) / 256), 256, 0, stream>>>(fc5w, W5h, 1000, 128, N5PAD, 128);
    }

    // ---- MLP: 672 -> 2048 -> 1024 -> 512 -> 128 -> 1000(pad 1024) ----
    const int mTiles = BATCH / 16;  // 16
    wmma_gemm_kernel<<<dim3(2048 / 64, mTiles), 32, 0, stream>>>(
        h0, W1h, fc1b, h1, nullptr, KPAD0, 2048, 2048, 1);
    wmma_gemm_kernel<<<dim3(1024 / 64, mTiles), 32, 0, stream>>>(
        h1, W2h, fc2b, h2, nullptr, 2048, 1024, 1024, 1);
    wmma_gemm_kernel<<<dim3(512 / 64, mTiles), 32, 0, stream>>>(
        h2, W3h, fc3b, h3, nullptr, 1024, 512, 512, 1);
    wmma_gemm_kernel<<<dim3(128 / 64, mTiles), 32, 0, stream>>>(
        h3, W4h, fc4b, h4, nullptr, 512, 128, 128, 1);
    wmma_gemm_kernel<<<dim3(N5PAD / 64, mTiles), 32, 0, stream>>>(
        h4, W5h, fc5b, nullptr, out, 128, 1000, 1000, 0);
}
